// SpaceBenderModel_61753039781993
// MI455X (gfx1250) — matrix operand order repaired and checked by reading, not compiled, not run on hardware
//
#include <hip/hip_runtime.h>
#include <hip/hip_bf16.h>

// ---------------------------------------------------------------------------
// SpaceBender model on MI455X (gfx1250): every GEMM goes through
// v_wmma_f32_16x16x32_f16 (wave32 WMMA), f32 accumulation, f32 LN/GELU.
// Output (524 MB f32) is the bandwidth bottleneck; all operands live in L2.
// ---------------------------------------------------------------------------

typedef _Float16 v16h __attribute__((ext_vector_type(16)));
typedef _Float16 v8h  __attribute__((ext_vector_type(8)));
typedef float    v8f  __attribute__((ext_vector_type(8)));

#define WMMA_F16(a, b, c) \
  __builtin_amdgcn_wmma_f32_16x16x32_f16(false, (a), false, (b), (short)0, (c), false, false)

// A-fragment (16x32 f16): lane l holds row M=l&15; halves 0..15 = K=kc..kc+15,
// where kc = k + 16*(l>=16).  Two 16B loads.
__device__ __forceinline__ v16h frag_a_ld(const _Float16* p) {
  v8h lo = *reinterpret_cast<const v8h*>(p);
  v8h hi = *reinterpret_cast<const v8h*>(p + 8);
  v16h f;
#pragma unroll
  for (int j = 0; j < 8; ++j) { f[j] = lo[j]; f[j + 8] = hi[j]; }
  return f;
}

// B-fragment (32x16 f16, stored transposed [N,K]): lane l holds col N=l&15;
// halves 0..15 = K = k + 16*(l>=16) .. +15.  One contiguous 32B load.
__device__ __forceinline__ v16h frag_b_ld(const _Float16* p) {
  return *reinterpret_cast<const v16h*>(p);
}

__device__ __forceinline__ v16h frag_zero() {
  v16h f;
#pragma unroll
  for (int j = 0; j < 16; ++j) f[j] = (_Float16)0.0f;
  return f;
}

__device__ __forceinline__ float gelu_exact(float v) {
  return 0.5f * v * (1.0f + erff(v * 0.70710678118654752f));
}

// ---------------------------------------------------------------------------
// Embedding gather + positional add -> x (f32) and xh (f16).
// ---------------------------------------------------------------------------
__global__ __launch_bounds__(256) void embed_kernel(
    const int* __restrict__ ids, const float* __restrict__ emb,
    const float* __restrict__ pos, float* __restrict__ xf,
    _Float16* __restrict__ xh, int D, int seq) {
  const int row = blockIdx.x;
  const int i = threadIdx.x;
  const int id = ids[row];
  const float v = emb[(size_t)id * D + i] + pos[(size_t)(row % seq) * D + i];
  xf[(size_t)row * D + i] = v;
  xh[(size_t)row * D + i] = (_Float16)v;
}

// ---------------------------------------------------------------------------
// Transpose-cast weights: W [K,N] f32 (row-major)  ->  WT [N,K] f16.
// ---------------------------------------------------------------------------
__global__ __launch_bounds__(256) void convT_kernel(
    const float* __restrict__ W, _Float16* __restrict__ WT, int K, int N) {
  const size_t idx = (size_t)blockIdx.x * 256 + threadIdx.x;
  if (idx >= (size_t)K * N) return;
  const int k = (int)(idx / N);
  const int n = (int)(idx % N);
  WT[(size_t)n * K + k] = (_Float16)W[idx];
}

// ---------------------------------------------------------------------------
// WMMA GEMM: C[M,N] = A[M,K] (f16, row-major) x BT[N,K]^T (+bias, +gelu).
// Block = 8 waves; wave (wm,wn) owns a 32x32 patch as 2x2 16x16 WMMA tiles.
// Block tile = 64(M) x 128(N).  M%64==0, N%128==0, K%32==0 guaranteed.
// ---------------------------------------------------------------------------
__global__ __launch_bounds__(256) void wmma_gemm_kernel(
    const _Float16* __restrict__ A, const _Float16* __restrict__ BT,
    const float* __restrict__ bias, float* __restrict__ Cf,
    _Float16* __restrict__ Ch, int M, int N, int K, int act) {
  const int lane = threadIdx.x & 31;
  const int wave = threadIdx.x >> 5;
  const int wm = wave >> 2, wn = wave & 3;
  const int m_base = blockIdx.y * 64 + wm * 32;
  const int n_base = blockIdx.x * 128 + wn * 32;
  const int r15 = lane & 15;
  const int khi8 = (lane >> 4) << 3;   // unused; A also uses khi16 (0 or 16)
  const int khi16 = (lane >> 4) << 4;  // B fragment K offset (0 or 16)

  v8f acc[2][2] = {};

  const _Float16* arow0 = A + (size_t)(m_base + r15) * K;
  const _Float16* arow1 = A + (size_t)(m_base + 16 + r15) * K;
  const _Float16* brow0 = BT + (size_t)(n_base + r15) * K;
  const _Float16* brow1 = BT + (size_t)(n_base + 16 + r15) * K;

  for (int k = 0; k < K; k += 32) {
    const int ka = k + khi16;
    const int kb = k + khi16;
    v16h a0 = frag_a_ld(arow0 + ka);
    v16h a1 = frag_a_ld(arow1 + ka);
    v16h b0 = frag_b_ld(brow0 + kb);
    v16h b1 = frag_b_ld(brow1 + kb);
    acc[0][0] = WMMA_F16(a0, b0, acc[0][0]);
    acc[0][1] = WMMA_F16(a0, b1, acc[0][1]);
    acc[1][0] = WMMA_F16(a1, b0, acc[1][0]);
    acc[1][1] = WMMA_F16(a1, b1, acc[1][1]);
  }

  const int hi8 = (lane >> 4) << 3;  // C/D: lanes 16-31 hold M rows +8
#pragma unroll
  for (int mi = 0; mi < 2; ++mi) {
#pragma unroll
    for (int ni = 0; ni < 2; ++ni) {
      const int col = n_base + ni * 16 + r15;
      const float bv = bias ? bias[col] : 0.0f;
#pragma unroll
      for (int r = 0; r < 8; ++r) {
        const int row = m_base + mi * 16 + hi8 + r;
        float v = acc[mi][ni][r] + bv;
        if (act) v = gelu_exact(v);
        if (Cf) Cf[(size_t)row * N + col] = v;
        if (Ch) Ch[(size_t)row * N + col] = (_Float16)v;
      }
    }
  }
}

// ---------------------------------------------------------------------------
// Local-mix GEMM: mixed[m] = sum_{j=0..2w} shift(xw, j-w)[m] @ W_j  (+ bias).
// Zero halo per batch (seq = 2048, power of two).  BT is [256, (2w+1)*256].
// ---------------------------------------------------------------------------
__global__ __launch_bounds__(256) void mix_gemm_kernel(
    const _Float16* __restrict__ A, const _Float16* __restrict__ BT,
    const float* __restrict__ bias, _Float16* __restrict__ Ch,
    int M, int N, int K, int J, int w, int seq) {
  const int lane = threadIdx.x & 31;
  const int wave = threadIdx.x >> 5;
  const int wm = wave >> 2, wn = wave & 3;
  const int m_base = blockIdx.y * 64 + wm * 32;
  const int n_base = blockIdx.x * 128 + wn * 32;
  const int r15 = lane & 15;
  const int khi8 = (lane >> 4) << 3;
  const int khi16 = (lane >> 4) << 4;
  const int Ktot = J * K;

  v8f acc[2][2] = {};

  const _Float16* brow0 = BT + (size_t)(n_base + r15) * Ktot;
  const _Float16* brow1 = BT + (size_t)(n_base + 16 + r15) * Ktot;

  for (int j = 0; j < J; ++j) {
    const int shift = j - w;
    const int g0 = m_base + r15;
    const int g1 = g0 + 16;
    const int p0 = (g0 & (seq - 1)) + shift;
    const int p1 = (g1 & (seq - 1)) + shift;
    const bool ok0 = (p0 >= 0) && (p0 < seq);
    const bool ok1 = (p1 >= 0) && (p1 < seq);
    const _Float16* arow0 = A + (size_t)(g0 + shift) * K;
    const _Float16* arow1 = A + (size_t)(g1 + shift) * K;
    const int joff = j * K;
    for (int k = 0; k < K; k += 32) {
      const int ka = k + khi16;
      const int kb = joff + k + khi16;
      v16h a0 = ok0 ? frag_a_ld(arow0 + ka) : frag_zero();
      v16h a1 = ok1 ? frag_a_ld(arow1 + ka) : frag_zero();
      v16h b0 = frag_b_ld(brow0 + kb);
      v16h b1 = frag_b_ld(brow1 + kb);
      acc[0][0] = WMMA_F16(a0, b0, acc[0][0]);
      acc[0][1] = WMMA_F16(a0, b1, acc[0][1]);
      acc[1][0] = WMMA_F16(a1, b0, acc[1][0]);
      acc[1][1] = WMMA_F16(a1, b1, acc[1][1]);
    }
  }

  const int hi8 = (lane >> 4) << 3;
#pragma unroll
  for (int mi = 0; mi < 2; ++mi) {
#pragma unroll
    for (int ni = 0; ni < 2; ++ni) {
      const int col = n_base + ni * 16 + r15;
      const float bv = bias[col];
#pragma unroll
      for (int r = 0; r < 8; ++r) {
        const int row = m_base + mi * 16 + hi8 + r;
        Ch[(size_t)row * N + col] = (_Float16)(acc[mi][ni][r] + bv);
      }
    }
  }
}

// ---------------------------------------------------------------------------
// x = LayerNorm(x + t) * s + b  (D = 256, one block per row, LDS reduction).
// Writes updated x (f32) and xh (f16).
// ---------------------------------------------------------------------------
__global__ __launch_bounds__(256) void ln_residual_kernel(
    float* __restrict__ xf, const float* __restrict__ t,
    const float* __restrict__ s, const float* __restrict__ b,
    _Float16* __restrict__ xh, int D) {
  __shared__ float red[256];
  const int row = blockIdx.x;
  const int i = threadIdx.x;
  const size_t idx = (size_t)row * D + i;

  const float v = xf[idx] + t[idx];

  red[i] = v;
  __syncthreads();
#pragma unroll
  for (int off = 128; off > 0; off >>= 1) {
    if (i < off) red[i] += red[i + off];
    __syncthreads();
  }
  const float mean = red[0] * (1.0f / 256.0f);
  __syncthreads();

  const float d = v - mean;
  red[i] = d * d;
  __syncthreads();
#pragma unroll
  for (int off = 128; off > 0; off >>= 1) {
    if (i < off) red[i] += red[i + off];
    __syncthreads();
  }
  const float var = red[0] * (1.0f / 256.0f);
  const float inv = rsqrtf(var + 1e-5f);

  const float o = d * inv * s[i] + b[i];
  xf[idx] = o;
  xh[idx] = (_Float16)o;
}

// ---------------------------------------------------------------------------
// Host-side orchestration.
// ---------------------------------------------------------------------------
extern "C" void kernel_launch(void* const* d_in, const int* in_sizes, int n_in,
                              void* d_out, int out_size, void* d_ws, size_t ws_size,
                              hipStream_t stream) {
  (void)in_sizes; (void)n_in; (void)out_size; (void)ws_size;

  constexpr int Vv = 32000, D = 256, Dff = 1024, SEQ = 2048, BN = 4096, NL = 4;

  const int*   ids    = (const int*)d_in[0];
  const float* emb    = (const float*)d_in[1];
  const float* pos    = (const float*)d_in[2];
  const float* head_w = (const float*)d_in[3];
  const float* head_b = (const float*)d_in[4];
  auto LW = [&](int l, int j) { return (const float*)d_in[5 + 12 * l + j]; };
  // layer tensor order: 0 warp, 1 mix_w, 2 mix_b, 3 unwarp, 4 ln1_s, 5 ln1_b,
  //                     6 ffn_w1, 7 ffn_b1, 8 ffn_w2, 9 ffn_b2, 10 ln2_s, 11 ln2_b

  // --- carve workspace (256B aligned) ---
  char* ws = (char*)d_ws;
  size_t off = 0;
  auto carve = [&](size_t bytes) -> void* {
    void* p = ws + off;
    off = (off + bytes + 255) & ~(size_t)255;
    return p;
  };
  float*    xf    = (float*)carve((size_t)BN * D * 4);
  _Float16* xh    = (_Float16*)carve((size_t)BN * D * 2);
  _Float16* xwh   = (_Float16*)carve((size_t)BN * D * 2);
  _Float16* mixh  = (_Float16*)carve((size_t)BN * D * 2);
  float*    tmp   = (float*)carve((size_t)BN * D * 4);
  _Float16* hh    = (_Float16*)carve((size_t)BN * Dff * 2);
  _Float16* wT    = (_Float16*)carve((size_t)Vv * D * 2);  // largest (head) weight

  auto convT = [&](const float* W, int K, int N) {
    const size_t n = (size_t)K * N;
    convT_kernel<<<(unsigned)((n + 255) / 256), 256, 0, stream>>>(W, wT, K, N);
  };
  auto gemm = [&](const _Float16* A, const float* bias, float* Cf, _Float16* Ch,
                  int M, int N, int K, int act) {
    dim3 g((unsigned)(N / 128), (unsigned)(M / 64));
    wmma_gemm_kernel<<<g, 256, 0, stream>>>(A, wT, bias, Cf, Ch, M, N, K, act);
  };

  // --- embedding ---
  embed_kernel<<<BN, 256, 0, stream>>>(ids, emb, pos, xf, xh, D, SEQ);

  // --- layers ---
  for (int l = 0; l < NL; ++l) {
    const int w = 2 + l, J = 2 * w + 1;

    // xw = x @ warp            (f16 result for the shifted-window mix)
    convT(LW(l, 0), D, D);
    gemm(xh, nullptr, nullptr, xwh, BN, D, D, 0);

    // mixed = feats @ mix_w + mix_b  ==  sum_j shift(xw, j-w) @ W_j + b
    convT(LW(l, 1), J * D, D);
    {
      dim3 g(D / 128, BN / 64);
      mix_gemm_kernel<<<g, 256, 0, stream>>>(xwh, wT, LW(l, 2), mixh,
                                             BN, D, D, J, w, SEQ);
    }

    // x = LN(x + mixed @ unwarp)
    convT(LW(l, 3), D, D);
    gemm(mixh, nullptr, tmp, nullptr, BN, D, D, 0);
    ln_residual_kernel<<<BN, 256, 0, stream>>>(xf, tmp, LW(l, 4), LW(l, 5), xh, D);

    // h = gelu(x @ ffn_w1 + b1)
    convT(LW(l, 6), D, Dff);
    gemm(xh, LW(l, 7), nullptr, hh, BN, Dff, D, 1);

    // x = LN(x + h @ ffn_w2 + b2)
    convT(LW(l, 8), Dff, D);
    gemm(hh, LW(l, 9), tmp, nullptr, BN, D, Dff, 0);
    ln_residual_kernel<<<BN, 256, 0, stream>>>(xf, tmp, LW(l, 10), LW(l, 11), xh, D);
  }

  // --- LM head: out = x @ head_w + head_b   (f32, streams 524 MB to HBM) ---
  convT(head_w, D, Vv);
  gemm(xh, head_b, (float*)d_out, nullptr, BN, Vv, D, 0);
}
